// MultiHeadAttention_46986942218365
// MI455X (gfx1250) — compile-verified, matrix operand order checked by reading, not run on hardware
//
#include <hip/hip_runtime.h>

// MI455X / gfx1250 multi-head attention:
//   B=4, S=2048, D=1024, H=16, HD=64
// Pipeline: QKV WMMA GEMMs (f32->f16, double-buffered LDS) -> flash attention
// (f16 WMMA, f32 softmax state, DPP row-max, WMMA row-sum, register-pipelined
//  K/V fragments) -> output WMMA GEMM (f16 -> f32 + bias).

typedef __attribute__((ext_vector_type(16))) _Float16 v16h;
typedef __attribute__((ext_vector_type(8)))  float    v8f;
typedef __attribute__((ext_vector_type(4)))  float    f32x4;
typedef __attribute__((ext_vector_type(4)))  _Float16 h16x4;

#define WMMA_F32_F16(a, b, c) \
  __builtin_amdgcn_wmma_f32_16x16x32_f16(false, (a), false, (b), (short)0, (c), false, false)

// Load a 16x32 f16 fragment in the CDNA5 WMMA A-operand layout from a
// row-major source (row stride `ld` in f16 elements).
//   lane 0-15 : row = lane,     kh = 0 : K = {0..7} U {16..23}
//   lane 16-31: row = lane-16,  kh = 8 : K = {8..15} U {24..31}
// The same layout applied to B^T rows produces the B operand, so x-rows (A)
// and W-rows / K-rows / V^T-rows (B) all load with this one helper.
__device__ __forceinline__ v16h load_frag(const _Float16* base, int ld, int lane) {
  const int row = lane & 15;
  const int kh  = (lane >> 4) << 3;  // 0 or 8
  const _Float16* p = base + row * ld + kh;
  v16h f;
  f32x4* fv = reinterpret_cast<f32x4*>(&f);
  fv[0] = *reinterpret_cast<const f32x4*>(p);       // K = kh..kh+7       -> VGPR0-3
  fv[1] = *reinterpret_cast<const f32x4*>(p + 16);  // K = 16+kh..16+kh+7 -> VGPR4-7
  return f;
}

// Max-reduce across the 16-lane row groups of the WMMA C layout using DPP
// ROW_XMASK swaps (0x160+d): pure VALU, co-executes with XDL WMMA, no LDS.
__device__ __forceinline__ float rowmax16_dpp(float x) {
  int t;
  t = __builtin_amdgcn_mov_dpp(__builtin_bit_cast(int, x), 0x161, 0xf, 0xf, false);  // xor 1
  x = fmaxf(x, __builtin_bit_cast(float, t));
  t = __builtin_amdgcn_mov_dpp(__builtin_bit_cast(int, x), 0x162, 0xf, 0xf, false);  // xor 2
  x = fmaxf(x, __builtin_bit_cast(float, t));
  t = __builtin_amdgcn_mov_dpp(__builtin_bit_cast(int, x), 0x164, 0xf, 0xf, false);  // xor 4
  x = fmaxf(x, __builtin_bit_cast(float, t));
  t = __builtin_amdgcn_mov_dpp(__builtin_bit_cast(int, x), 0x168, 0xf, 0xf, false);  // xor 8
  x = fmaxf(x, __builtin_bit_cast(float, t));
  return x;
}

// ---------------------------------------------------------------------------
// Tiled GEMM:  Out = X @ W^T + bias     (X: [M,Kd], W: [N,Kd])
// Block tile 128x128, 8 waves: waveM in 0..3 (32 rows), waveN in 0..1 (64 cols)
// Double-buffered LDS: one barrier per k-step; global fetch of tile k+1
// overlaps WMMA compute of tile k.
// OUTMODE 0: f16 out, per-head layout [B,H,S,HD]     (Q, K)
// OUTMODE 1: f16 out, per-head transposed [B,H,HD,S] (V)
// OUTMODE 2: f32 out, row-major [M,N]                (final projection)
// ---------------------------------------------------------------------------
template <bool XF16>
__device__ __forceinline__ void gstage_load(const void* Xv, const float* W, int Kd,
                                            int mBase, int nBase, int k0,
                                            int ldRow, int ldHalf, f32x4* xr, f32x4* wr) {
  if (XF16) {
    const _Float16* Xp = ((const _Float16*)Xv) + (size_t)(mBase + ldRow) * Kd + k0 + ldHalf * 16;
    xr[0] = *reinterpret_cast<const f32x4*>(Xp);
    xr[1] = *reinterpret_cast<const f32x4*>(Xp + 8);
  } else {
    const float* Xp = ((const float*)Xv) + (size_t)(mBase + ldRow) * Kd + k0 + ldHalf * 16;
#pragma unroll
    for (int i = 0; i < 4; ++i) xr[i] = *reinterpret_cast<const f32x4*>(Xp + 4 * i);
  }
  const float* Wp = W + (size_t)(nBase + ldRow) * Kd + k0 + ldHalf * 16;
#pragma unroll
  for (int i = 0; i < 4; ++i) wr[i] = *reinterpret_cast<const f32x4*>(Wp + 4 * i);
}

template <bool XF16>
__device__ __forceinline__ void gstage_store(_Float16 (&Xs)[128][40], _Float16 (&Ws)[128][40],
                                             const f32x4* xr, const f32x4* wr,
                                             int ldRow, int ldHalf) {
  if (XF16) {
    *reinterpret_cast<f32x4*>(&Xs[ldRow][ldHalf * 16])     = xr[0];
    *reinterpret_cast<f32x4*>(&Xs[ldRow][ldHalf * 16 + 8]) = xr[1];
  } else {
#pragma unroll
    for (int i = 0; i < 4; ++i) {
      h16x4 h = { (_Float16)xr[i][0], (_Float16)xr[i][1], (_Float16)xr[i][2], (_Float16)xr[i][3] };
      *reinterpret_cast<h16x4*>(&Xs[ldRow][ldHalf * 16 + 4 * i]) = h;
    }
  }
#pragma unroll
  for (int i = 0; i < 4; ++i) {
    h16x4 h = { (_Float16)wr[i][0], (_Float16)wr[i][1], (_Float16)wr[i][2], (_Float16)wr[i][3] };
    *reinterpret_cast<h16x4*>(&Ws[ldRow][ldHalf * 16 + 4 * i]) = h;
  }
}

template <bool XF16, int OUTMODE>
__global__ __launch_bounds__(256) void gemm_xwt(
    const void* __restrict__ Xv, const float* __restrict__ W,
    const float* __restrict__ bias, void* __restrict__ Out,
    int M, int Kd, int N) {
  __shared__ _Float16 Xs[2][128][40];  // +8 pad -> conflict-free b128 frag loads
  __shared__ _Float16 Ws[2][128][40];

  const int tid   = threadIdx.x;
  const int lane  = tid & 31;
  const int wave  = tid >> 5;
  const int waveM = wave & 3;
  const int waveN = wave >> 2;
  const int mBase = blockIdx.y * 128;
  const int nBase = blockIdx.x * 128;

  v8f acc[2][4];
#pragma unroll
  for (int i = 0; i < 2; ++i)
#pragma unroll
    for (int j = 0; j < 4; ++j) acc[i][j] = {};

  const int ldRow  = tid >> 1;  // 0..127
  const int ldHalf = tid & 1;   // 16-col half

  f32x4 xr[4], wr[4];
  gstage_load<XF16>(Xv, W, Kd, mBase, nBase, 0, ldRow, ldHalf, xr, wr);
  gstage_store<XF16>(Xs[0], Ws[0], xr, wr, ldRow, ldHalf);

  const int nT = Kd / 32;
  int cur = 0;
  for (int kt = 0; kt < nT; ++kt) {
    __syncthreads();  // stores to buffer `cur` visible; prior reads of cur^1 done
    const bool more = (kt + 1 < nT);
    if (more) gstage_load<XF16>(Xv, W, Kd, mBase, nBase, (kt + 1) * 32, ldRow, ldHalf, xr, wr);

    v16h a[2], b[4];
#pragma unroll
    for (int tm = 0; tm < 2; ++tm)
      a[tm] = load_frag(&Xs[cur][waveM * 32 + tm * 16][0], 40, lane);
#pragma unroll
    for (int tn = 0; tn < 4; ++tn)
      b[tn] = load_frag(&Ws[cur][waveN * 64 + tn * 16][0], 40, lane);
#pragma unroll
    for (int tm = 0; tm < 2; ++tm)
#pragma unroll
      for (int tn = 0; tn < 4; ++tn)
        acc[tm][tn] = WMMA_F32_F16(a[tm], b[tn], acc[tm][tn]);

    if (more) gstage_store<XF16>(Xs[cur ^ 1], Ws[cur ^ 1], xr, wr, ldRow, ldHalf);
    cur ^= 1;
  }

  // Epilogue. C layout: reg r, lanes 0-15 -> row r, lanes 16-31 -> row 8+r,
  // column = lane & 15.
  const int ln = lane & 15;
  const int mh = (lane < 16) ? 0 : 8;
#pragma unroll
  for (int tm = 0; tm < 2; ++tm) {
#pragma unroll
    for (int tn = 0; tn < 4; ++tn) {
      const int nCol = nBase + waveN * 64 + tn * 16 + ln;
      const float bv = bias[nCol];
#pragma unroll
      for (int r = 0; r < 8; ++r) {
        const int mRow = mBase + waveM * 32 + tm * 16 + mh + r;
        const float val = acc[tm][tn][r] + bv;
        if (OUTMODE == 2) {
          ((float*)Out)[(size_t)mRow * N + nCol] = val;
        } else {
          const int bidx = mRow >> 11, s = mRow & 2047;  // S = 2048
          const int h = nCol >> 6, hd = nCol & 63;       // HD = 64
          const int bh = bidx * 16 + h;                  // H = 16
          const size_t addr = (OUTMODE == 0)
                                  ? ((size_t)(bh * 2048 + s) * 64 + hd)   // [B,H,S,HD]
                                  : ((size_t)(bh * 64 + hd) * 2048 + s);  // [B,H,HD,S]
          ((_Float16*)Out)[addr] = (_Float16)val;
        }
      }
    }
  }
}

// ---------------------------------------------------------------------------
// Flash attention: one wave owns 16 query rows; 8 waves -> 128 q rows / block.
// grid = (S/128, B*H).  Q,K: [B,H,S,64] f16.  Vt: [B,H,64,S] f16.
// X2 out: [B,S,D] f16 (heads merged) for the final projection.
// K/V fragments are register-pipelined (ping-pong, 2x-unrolled loop) so each
// chunk's global loads are issued a full chunk of compute before first use.
// ---------------------------------------------------------------------------
struct AttnState {
  v8f O[4];
  float mrow[8];
  float lrow[8];
};

__device__ __forceinline__ void load_kv(const _Float16* Kh, const _Float16* Vh,
                                        int j, int lane, v16h* kb, v16h* vb) {
  kb[0] = load_frag(Kh + (size_t)j * 64,             64, lane);
  kb[1] = load_frag(Kh + (size_t)j * 64 + 32,        64, lane);
  kb[2] = load_frag(Kh + (size_t)(j + 16) * 64,      64, lane);
  kb[3] = load_frag(Kh + (size_t)(j + 16) * 64 + 32, 64, lane);
#pragma unroll
  for (int t = 0; t < 4; ++t)
    vb[t] = load_frag(Vh + (size_t)(t * 16) * 2048 + j, 2048, lane);
}

__device__ __forceinline__ void attn_chunk(const v16h& qa0, const v16h& qa1,
                                           const v16h* kb, const v16h* vb,
                                           const v16h& ones, AttnState& st,
                                           _Float16 (&P)[16][32],
                                           int lane, int ln, int mh) {
  // scores: S[16 q x 32 keys] = Q(16x64) @ K^T, two 16x16 n-tiles
  v8f s0 = {}, s1 = {};
  s0 = WMMA_F32_F16(qa0, kb[0], s0);
  s0 = WMMA_F32_F16(qa1, kb[1], s0);
  s1 = WMMA_F32_F16(qa0, kb[2], s1);
  s1 = WMMA_F32_F16(qa1, kb[3], s1);

  // Online softmax: row max (DPP), rescale factor, exponentials.
  float al[8];
#pragma unroll
  for (int r = 0; r < 8; ++r) {
    const float a = s0[r] * 0.125f;  // 1/sqrt(HD)
    const float b = s1[r] * 0.125f;
    const float mx = rowmax16_dpp(fmaxf(a, b));
    const float mnew = fmaxf(st.mrow[r], mx);
    al[r] = __expf(st.mrow[r] - mnew);
    st.mrow[r] = mnew;
    s0[r] = __expf(a - mnew);
    s1[r] = __expf(b - mnew);
  }

  // C-layout -> A-layout transpose of P through wave-private LDS
  // (same-wave DS ops are in order; no barrier needed).
#pragma unroll
  for (int r = 0; r < 8; ++r) {
    P[mh + r][ln]      = (_Float16)s0[r];
    P[mh + r][16 + ln] = (_Float16)s1[r];
  }
  const v16h pa = load_frag(&P[0][0], 32, lane);

  // Row sums of P via WMMA against ones; update l with rescale.
  v8f rsum = {};
  rsum = WMMA_F32_F16(pa, ones, rsum);
#pragma unroll
  for (int r = 0; r < 8; ++r) st.lrow[r] = st.lrow[r] * al[r] + rsum[r];

  // O = diag(alpha) * O + P @ V_chunk   (N = 64 -> 4 tiles, K = 32)
#pragma unroll
  for (int t = 0; t < 4; ++t) {
#pragma unroll
    for (int r = 0; r < 8; ++r) st.O[t][r] *= al[r];
    st.O[t] = WMMA_F32_F16(pa, vb[t], st.O[t]);
  }
}

__global__ __launch_bounds__(256) void attn_kernel(
    const _Float16* __restrict__ Q, const _Float16* __restrict__ K,
    const _Float16* __restrict__ Vt, _Float16* __restrict__ X2) {
  __shared__ _Float16 Pls[8][16][32];  // wave-private P transpose staging (8 KB)

  const int tid  = threadIdx.x;
  const int lane = tid & 31;
  const int wave = tid >> 5;
  const int bh   = blockIdx.y;
  const int qBase = blockIdx.x * 128 + wave * 16;

  const _Float16* Qh = Q  + (size_t)bh * 2048 * 64;
  const _Float16* Kh = K  + (size_t)bh * 2048 * 64;
  const _Float16* Vh = Vt + (size_t)bh * 64 * 2048;

  // Q fragments for this wave's 16 rows (K = 0..31 and 32..63), kept resident.
  const v16h qa0 = load_frag(Qh + (size_t)qBase * 64,      64, lane);
  const v16h qa1 = load_frag(Qh + (size_t)qBase * 64 + 32, 64, lane);

  // All-ones B fragment: rowsum(P) = P @ ones via one WMMA (every lane of a
  // row group receives that row's sum in the C layout).
  v16h ones;
#pragma unroll
  for (int i = 0; i < 16; ++i) ones[i] = (_Float16)1.0f;

  AttnState st;
#pragma unroll
  for (int t = 0; t < 4; ++t) st.O[t] = {};
#pragma unroll
  for (int r = 0; r < 8; ++r) { st.mrow[r] = -3.0e38f; st.lrow[r] = 0.0f; }

  const int ln = lane & 15;
  const int mh = (lane < 16) ? 0 : 8;
  auto& Pw = Pls[wave];

  // Ping-pong register pipeline over 64-key super-chunks (2 x 32-key chunks).
  v16h kbA[4], vbA[4], kbB[4], vbB[4];
  load_kv(Kh, Vh, 0, lane, kbA, vbA);
  for (int j = 0; j < 2048; j += 64) {
    load_kv(Kh, Vh, j + 32, lane, kbB, vbB);                 // prefetch chunk j+32
    attn_chunk(qa0, qa1, kbA, vbA, ones, st, Pw, lane, ln, mh);  // compute chunk j
    const int jn = (j + 64 < 2048) ? (j + 64) : 2016;        // clamp (last reload unused)
    load_kv(Kh, Vh, jn, lane, kbA, vbA);                     // prefetch chunk j+64
    attn_chunk(qa0, qa1, kbB, vbB, ones, st, Pw, lane, ln, mh); // compute chunk j+32
  }

  // Normalize and write merged-head output [B,S,D] as f16.
  const int bidx = bh >> 4, h = bh & 15;
#pragma unroll
  for (int t = 0; t < 4; ++t) {
#pragma unroll
    for (int r = 0; r < 8; ++r) {
      const int s = qBase + mh + r;
      const float val = st.O[t][r] / st.lrow[r];
      X2[(size_t)(bidx * 2048 + s) * 1024 + h * 64 + t * 16 + ln] = (_Float16)val;
    }
  }
}

// ---------------------------------------------------------------------------
extern "C" void kernel_launch(void* const* d_in, const int* in_sizes, int n_in,
                              void* d_out, int out_size, void* d_ws, size_t ws_size,
                              hipStream_t stream) {
  (void)in_sizes; (void)n_in; (void)out_size; (void)ws_size;
  const float* query = (const float*)d_in[0];
  const float* key   = (const float*)d_in[1];
  const float* value = (const float*)d_in[2];
  const float* Wq = (const float*)d_in[3];
  const float* bq = (const float*)d_in[4];
  const float* Wk = (const float*)d_in[5];
  const float* bk = (const float*)d_in[6];
  const float* Wv = (const float*)d_in[7];
  const float* bv = (const float*)d_in[8];
  const float* Wo = (const float*)d_in[9];
  const float* bo = (const float*)d_in[10];

  const int M = 8192, D = 1024;        // M = B*S
  const size_t elems = (size_t)M * D;  // 8,388,608 f16 per buffer
  _Float16* Qbuf = (_Float16*)d_ws;
  _Float16* Kbuf = Qbuf + elems;
  _Float16* Vtb  = Kbuf + elems;
  _Float16* X2   = Vtb + elems;        // 64 MB total workspace

  dim3 blk(256);
  dim3 gGemm(D / 128, M / 128);        // (8, 64)

  gemm_xwt<false, 0><<<gGemm, blk, 0, stream>>>(query, Wq, bq, Qbuf, M, D, D);
  gemm_xwt<false, 0><<<gGemm, blk, 0, stream>>>(key,   Wk, bk, Kbuf, M, D, D);
  gemm_xwt<false, 1><<<gGemm, blk, 0, stream>>>(value, Wv, bv, Vtb,  M, D, D);

  attn_kernel<<<dim3(2048 / 128, 4 * 16), blk, 0, stream>>>(Qbuf, Kbuf, Vtb, X2);

  gemm_xwt<true, 2><<<gGemm, blk, 0, stream>>>(X2, Wo, bo, d_out, M, D, D);
}
